// iComformer_74277164417077
// MI455X (gfx1250) — matrix operand order repaired and checked by reading, not compiled, not run on hardware
//
#include <hip/hip_runtime.h>

// iComformer forward for gfx1250 (MI455X).
// All dense GEMMs use v_wmma_f32_16x16x32_bf16 (A/B fp32 -> bf16 via native cvt,
// fp32 accum). GEMM: 128x64 tile, BK=64, 2x2 accumulators/wave, double-buffered
// LDS, vectorized b128 staging, templated A-source -> branch-free inner loop.
// Workspace requirement: ~3.0 GB.

#define C_H   256
#define NB    5000
#define EB    100000
#define E3    300000
#define NGR   32
#define AF    92

typedef __bf16 bf16;
typedef __attribute__((ext_vector_type(4)))  bf16  v4bf;
typedef __attribute__((ext_vector_type(8)))  bf16  v8bf;
typedef __attribute__((ext_vector_type(16))) bf16  v16bf;
typedef __attribute__((ext_vector_type(8)))  float v8f;

// ---------------------------------------------------------------------------
// Tiled WMMA GEMM:  C[M,N] (op)= act( A' @ W^T + biasScale*bias )
//   MODE 0: A' row m = A[m]            (ldA)
//   MODE 1: A' row m = A[idxA[m]]
//   MODE 2: A' row m = A[(m/aDiv)*aMul + aAdd]
//   MODE 3: A'[m,k] = exp(-gamma*(dvec[m]-(vmin+k*step))^2)   (fused RBF)
//   W is [N x ldW] row-major (row n contiguous in K) -> computes x @ W^T.
//   Output row m written at C[(m*oMul+oAdd)*ldC + n].
//   acc: 0 overwrite, 1 add to existing C.  act: 0 none, 1 silu, 2 softplus.
// Block: 256 threads (8 wave32). Tile 128x64, BK=64, double-buffered LDS.
// Each wave owns a 32x32 output patch = 2x2 WMMA accumulators.
// ---------------------------------------------------------------------------
#define LDT 72   // LDS row stride in halves (144 B = 9*16)

template <int MODE>
__global__ __launch_bounds__(256)
void gemm_bf16_k(const float* __restrict__ A, const float* __restrict__ dvec,
                 const int* __restrict__ idxA, int aDiv, int aMul, int aAdd, int ldA,
                 const float* __restrict__ W, int ldW,
                 const float* __restrict__ bias, float biasScale,
                 float* __restrict__ Cmat, int oMul, int oAdd, int ldC,
                 int M, int N, int K, int acc, int act,
                 float vmin, float step, float gamma)
{
  __shared__ __align__(16) bf16 sA[2][128 * LDT];
  __shared__ __align__(16) bf16 sB[2][64 * LDT];

  const int tid  = threadIdx.x;
  const int lane = tid & 31;
  const int wave = tid >> 5;
  const int m0   = blockIdx.x * 128;
  const int n0   = blockIdx.y * 64;

  // ---- staging assignment ----
  // A: 128 rows x 64 cols; thread -> row tid>>1, 32-col half.
  const int sra  = tid >> 1;
  const int sega = (tid & 1) << 5;          // 0 or 32
  const int gma  = m0 + sra;
  // B: 64 rows x 64 cols; thread -> row tid>>2, 16-col quarter.
  const int srb  = tid >> 2;
  const int segb = (tid & 3) << 4;          // 0,16,32,48
  const int gnb  = n0 + srb;

  long arow = 0;
  if (MODE == 0) arow = gma;
  else if (MODE == 1) arow = (gma < M) ? (long)idxA[gma] : 0;
  else if (MODE == 2) arow = (long)(gma / aDiv) * aMul + aAdd;
  const float* Aptr = (MODE == 3) ? (const float*)0 : (A + arow * (long)ldA);
  const float  dval = (MODE == 3) ? ((gma < M) ? dvec[gma] : 0.f) : 0.f;
  const float* Wptr = W + (long)gnb * ldW;

  float fa[32], fb[16];

  auto loadA = [&](int k0) {
#pragma unroll
    for (int j = 0; j < 32; j += 4) {
      int gk = k0 + sega + j;
      if (MODE == 3) {
#pragma unroll
        for (int u = 0; u < 4; ++u) {
          float t = dval - (vmin + step * (float)(gk + u));
          fa[j + u] = __expf(-gamma * t * t);
        }
      } else if (gma < M && gk + 3 < K) {
        float4 v = *(const float4*)(Aptr + gk);
        fa[j] = v.x; fa[j + 1] = v.y; fa[j + 2] = v.z; fa[j + 3] = v.w;
      } else {
#pragma unroll
        for (int u = 0; u < 4; ++u)
          fa[j + u] = (gma < M && gk + u < K) ? Aptr[gk + u] : 0.f;
      }
    }
  };
  auto loadB = [&](int k0) {
#pragma unroll
    for (int j = 0; j < 16; j += 4) {
      int gk = k0 + segb + j;
      if (gnb < N && gk + 3 < K) {
        float4 v = *(const float4*)(Wptr + gk);
        fb[j] = v.x; fb[j + 1] = v.y; fb[j + 2] = v.z; fb[j + 3] = v.w;
      } else {
#pragma unroll
        for (int u = 0; u < 4; ++u)
          fb[j + u] = (gnb < N && gk + u < K) ? Wptr[gk + u] : 0.f;
      }
    }
  };
  auto stage = [&](int buf) {
    bf16* pA = &sA[buf][sra * LDT + sega];
#pragma unroll
    for (int j = 0; j < 32; j += 4) {
      v4bf t;
#pragma unroll
      for (int u = 0; u < 4; ++u) t[u] = (bf16)fa[j + u];
      *(v4bf*)(pA + j) = t;
    }
    bf16* pB = &sB[buf][srb * LDT + segb];
#pragma unroll
    for (int j = 0; j < 16; j += 4) {
      v4bf t;
#pragma unroll
      for (int u = 0; u < 4; ++u) t[u] = (bf16)fb[j + u];
      *(v4bf*)(pB + j) = t;
    }
  };

  // ---- per-wave fragment geometry: wave owns M-tiles {mtp,mtp+1} x N-tiles {ntp,ntp+1}
  const int mtp   = (wave & 3) << 1;    // 0,2,4,6  (of 8 M-tiles)
  const int ntp   = (wave >> 2) << 1;   // 0,2      (of 4 N-tiles)
  const int afr0  = mtp * 16 + (lane & 15);
  const int afr1  = afr0 + 16;
  const int aoff  = (lane < 16) ? 0 : 16;
  const int brow0 = ntp * 16 + (lane & 15);
  const int brow1 = brow0 + 16;
  const int boff  = (lane < 16) ? 0 : 16;

  auto fragA = [&](const bf16* base, int row, int ks) {
    const bf16* p = base + row * LDT + ks + aoff;
    v8bf lo = *(const v8bf*)p;
    v8bf hi = *(const v8bf*)(p + 8);
    v16bf f;
#pragma unroll
    for (int i = 0; i < 8; ++i) { f[i] = lo[i]; f[i + 8] = hi[i]; }
    return f;
  };
  auto fragB = [&](const bf16* base, int row, int ks) {
    const bf16* p = base + row * LDT + ks + boff;
    v8bf lo = *(const v8bf*)p;
    v8bf hi = *(const v8bf*)(p + 8);
    v16bf f;
#pragma unroll
    for (int i = 0; i < 8; ++i) { f[i] = lo[i]; f[i + 8] = hi[i]; }
    return f;
  };

  v8f accs[2][2];
#pragma unroll
  for (int mi = 0; mi < 2; ++mi)
#pragma unroll
    for (int ni = 0; ni < 2; ++ni)
      accs[mi][ni] = v8f{0.f, 0.f, 0.f, 0.f, 0.f, 0.f, 0.f, 0.f};

  const int nIter = (K + 63) >> 6;
  loadA(0); loadB(0);
  stage(0);

  for (int it = 0; it < nIter; ++it) {
    __syncthreads();
    if (it + 1 < nIter) { loadA((it + 1) << 6); loadB((it + 1) << 6); }
    const bf16* bA = sA[it & 1];
    const bf16* bB = sB[it & 1];
#pragma unroll
    for (int ks = 0; ks < 64; ks += 32) {
      v16bf a0 = fragA(bA, afr0, ks);
      v16bf a1 = fragA(bA, afr1, ks);
      v16bf b0 = fragB(bB, brow0, ks);
      v16bf b1 = fragB(bB, brow1, ks);
      accs[0][0] = __builtin_amdgcn_wmma_f32_16x16x32_bf16(false, a0, false, b0,
                                                           (short)0, accs[0][0], false, false);
      accs[0][1] = __builtin_amdgcn_wmma_f32_16x16x32_bf16(false, a0, false, b1,
                                                           (short)0, accs[0][1], false, false);
      accs[1][0] = __builtin_amdgcn_wmma_f32_16x16x32_bf16(false, a1, false, b0,
                                                           (short)0, accs[1][0], false, false);
      accs[1][1] = __builtin_amdgcn_wmma_f32_16x16x32_bf16(false, a1, false, b1,
                                                           (short)0, accs[1][1], false, false);
    }
    if (it + 1 < nIter) stage((it + 1) & 1);
  }

  // ---- epilogue: C/D layout VGPR r -> M = r (+8 for hi half-wave) ----
#pragma unroll
  for (int mi = 0; mi < 2; ++mi) {
    const int mb = m0 + (mtp + mi) * 16 + ((lane < 16) ? 0 : 8);
#pragma unroll
    for (int ni = 0; ni < 2; ++ni) {
      const int nc = n0 + (ntp + ni) * 16 + (lane & 15);
#pragma unroll
      for (int r = 0; r < 8; ++r) {
        int om = mb + r;
        if (om < M) {
          long off = ((long)om * oMul + oAdd) * ldC + nc;
          float v = accs[mi][ni][r];
          if (acc) v += Cmat[off];
          if (bias) v += biasScale * bias[nc];
          if (act == 1) v = v / (1.f + __expf(-v));
          else if (act == 2) v = (v > 20.f) ? v : log1pf(__expf(v));
          Cmat[off] = v;
        }
      }
    }
  }
}

// ---------------------------------------------------------------------------
// Elementwise / reduction kernels
// ---------------------------------------------------------------------------
__global__ void zero_k(float* p, long n) {
  long i = (long)blockIdx.x * 256 + threadIdx.x;
  if (i < n) p[i] = 0.f;
}

__global__ void geom_k(const float* __restrict__ attr, const float* __restrict__ nei,
                       float* __restrict__ d_ef, float* __restrict__ d_nl,
                       float* __restrict__ d_cos) {
  int e = blockIdx.x * 256 + threadIdx.x;
  if (e >= EB) return;
  float a0 = attr[e * 3], a1 = attr[e * 3 + 1], a2 = attr[e * 3 + 2];
  float an = sqrtf(a0 * a0 + a1 * a1 + a2 * a2);
  d_ef[e] = -0.75f / an;
#pragma unroll
  for (int i = 0; i < 3; ++i) {
    const float* p = nei + (long)e * 9 + i * 3;
    float n0 = p[0], n1 = p[1], n2 = p[2];
    float nn = sqrtf(n0 * n0 + n1 * n1 + n2 * n2);
    d_nl[e * 3 + i] = -0.75f / nn;
    float cs = (n0 * a0 + n1 * a1 + n2 * a2) / (nn * an);
    d_cos[e * 3 + i] = fminf(1.f, fmaxf(-1.f, cs));
  }
}

// alpha = q[row] * kj / 16, in place in kj. row = idx[m] (conv) or m/3 (edge).
__global__ void alpha_k(float* __restrict__ kj, const float* __restrict__ q,
                        const int* __restrict__ idx, long M) {
  long i = (long)blockIdx.x * 256 + threadIdx.x;
  if (i >= M * C_H) return;
  long m = i >> 8;
  int  c = (int)(i & 255);
  long qr = idx ? (long)idx[m] : (m / 3);
  kj[i] = q[qr * C_H + c] * kj[i] * 0.0625f;
}

// per-channel sum / sumsq -> stats[0..255], stats[256..511]
__global__ void colstats_k(const float* __restrict__ X, long M, float* __restrict__ stats) {
  int c = threadIdx.x;
  long r0 = (long)blockIdx.x * 64;
  long r1 = r0 + 64; if (r1 > M) r1 = M;
  float s = 0.f, ss = 0.f;
  for (long r = r0; r < r1; ++r) {
    float v = X[r * C_H + c];
    s += v; ss += v * v;
  }
  atomicAdd(&stats[c], s);
  atomicAdd(&stats[C_H + c], ss);
}

// msg *= sigmoid(batchnorm(alpha))
__global__ void gate_k(float* __restrict__ msg, const float* __restrict__ alpha,
                       const float* __restrict__ stats, const float* __restrict__ g,
                       const float* __restrict__ b, long M) {
  long i = (long)blockIdx.x * 256 + threadIdx.x;
  if (i >= M * C_H) return;
  int c = (int)(i & 255);
  float inv_m = 1.f / (float)M;
  float mean = stats[c] * inv_m;
  float var  = stats[C_H + c] * inv_m - mean * mean;
  float a = g[c] * (alpha[i] - mean) * rsqrtf(var + 1e-5f) + b[c];
  msg[i] *= 1.f / (1.f + __expf(-a));
}

__global__ void scatter_k(float* __restrict__ agg, const float* __restrict__ msg,
                          const int* __restrict__ dst, long M) {
  long i = (long)blockIdx.x * 256 + threadIdx.x;
  if (i >= M * C_H) return;
  long m = i >> 8;
  atomicAdd(&agg[(long)dst[m] * C_H + (i & 255)], msg[i]);
}

// out = softplus(x + batchnorm(y))
__global__ void bnres_k(float* __restrict__ out, const float* __restrict__ x,
                        const float* __restrict__ y, const float* __restrict__ stats,
                        const float* __restrict__ g, const float* __restrict__ b, long M) {
  long i = (long)blockIdx.x * 256 + threadIdx.x;
  if (i >= M * C_H) return;
  int c = (int)(i & 255);
  float inv_m = 1.f / (float)M;
  float mean = stats[c] * inv_m;
  float var  = stats[C_H + c] * inv_m - mean * mean;
  float v = x[i] + g[c] * (y[i] - mean) * rsqrtf(var + 1e-5f) + b[c];
  out[i] = (v > 20.f) ? v : log1pf(__expf(v));
}

__global__ void sum3_k(float* __restrict__ outs, const float* __restrict__ msg3) {
  long i = (long)blockIdx.x * 256 + threadIdx.x;
  if (i >= (long)EB * C_H) return;
  long e = i >> 8;
  int  c = (int)(i & 255);
  long b0 = (3 * e) << 8;
  outs[i] = msg3[b0 + c] + msg3[b0 + C_H + c] + msg3[b0 + 2 * C_H + c];
}

__global__ void count_k(float* counts, const int* __restrict__ batch) {
  int n = blockIdx.x * 256 + threadIdx.x;
  if (n < NB) atomicAdd(&counts[batch[n]], 1.f);
}

__global__ void pooladd_k(float* __restrict__ pooled, const float* __restrict__ nf,
                          const int* __restrict__ batch) {
  long i = (long)blockIdx.x * 256 + threadIdx.x;
  if (i >= (long)NB * C_H) return;
  long n = i >> 8;
  atomicAdd(&pooled[(long)batch[n] * C_H + (i & 255)], nf[i]);
}

__global__ void pooldiv_k(float* pooled, const float* counts) {
  int i = blockIdx.x * 256 + threadIdx.x;
  if (i < NGR * C_H) pooled[i] /= fmaxf(counts[i >> 8], 1.f);
}

__global__ void fcout_k(const float* __restrict__ feats, const float* __restrict__ w,
                        const float* __restrict__ b, float* __restrict__ out) {
  int g = threadIdx.x;
  if (g >= NGR) return;
  float s = 0.f;
  for (int c = 0; c < C_H; ++c) s += feats[g * C_H + c] * w[c];
  out[g] = s + b[0];
}

// ---------------------------------------------------------------------------
// Host-side orchestration
// ---------------------------------------------------------------------------
namespace {

struct Lin { const float* w; const float* b; };
struct Mlp { Lin l1, l2; };
struct BN  { const float* g; const float* b; };
struct ConvP { BN bn, bn_att; Lin concat, edge, k; Mlp key, msg; Lin q, v; };
struct EdgeP { BN bn, bn_att; Lin concat; const float* edge_w; Lin k, ke[3];
               Mlp key, msg; Lin q, v; Lin ve[3]; };

struct WS {
  float *nf0, *nf1, *ef0, *ef1, *nl, *na;
  float *d_ef, *d_nl, *d_cos, *stats, *pooled, *counts, *feats;
  float *q, *k, *v, *ea, *h, *kj, *msg, *agg, *aggc;                 // conv view
  float *qe, *ke, *ve, *exy, *ky, *h3, *kk, *msg3, *outs, *outc;     // edge view
};

// mode: 0 plain, 1 gather(idx), 2 affine(aDiv/aMul/aAdd), 3 RBF
inline void gemm(hipStream_t s, int mode,
                 const float* A, const float* dvec, const int* idxA,
                 int aDiv, int aMul, int aAdd, int ldA,
                 const float* W, int ldW, const float* bias, float bscale,
                 float* C, int oMul, int oAdd, int ldC,
                 int M, int N, int K, int acc, int act,
                 float vmin = 0.f, float step = 0.f, float gamma = 0.f) {
  dim3 g((unsigned)((M + 127) / 128), (unsigned)(N / 64));
  dim3 b(256);
  switch (mode) {
    case 0: gemm_bf16_k<0><<<g, b, 0, s>>>(A, dvec, idxA, aDiv, aMul, aAdd, ldA, W, ldW,
              bias, bscale, C, oMul, oAdd, ldC, M, N, K, acc, act, vmin, step, gamma); break;
    case 1: gemm_bf16_k<1><<<g, b, 0, s>>>(A, dvec, idxA, aDiv, aMul, aAdd, ldA, W, ldW,
              bias, bscale, C, oMul, oAdd, ldC, M, N, K, acc, act, vmin, step, gamma); break;
    case 2: gemm_bf16_k<2><<<g, b, 0, s>>>(A, dvec, idxA, aDiv, aMul, aAdd, ldA, W, ldW,
              bias, bscale, C, oMul, oAdd, ldC, M, N, K, acc, act, vmin, step, gamma); break;
    default: gemm_bf16_k<3><<<g, b, 0, s>>>(A, dvec, idxA, aDiv, aMul, aAdd, ldA, W, ldW,
              bias, bscale, C, oMul, oAdd, ldC, M, N, K, acc, act, vmin, step, gamma); break;
  }
}

inline unsigned blk(long n) { return (unsigned)((n + 255) / 256); }

void run_conv(hipStream_t s, const ConvP& P, const float* xin, float* xout,
              const float* e, const int* src, const int* dst, WS& w) {
  gemm(s, 0, xin, 0, 0, 1, 1, 0, C_H, P.q.w, C_H, P.q.b, 1.f, w.q, 1, 0, C_H, NB, C_H, C_H, 0, 0);
  gemm(s, 0, xin, 0, 0, 1, 1, 0, C_H, P.k.w, C_H, P.k.b, 1.f, w.k, 1, 0, C_H, NB, C_H, C_H, 0, 0);
  gemm(s, 0, xin, 0, 0, 1, 1, 0, C_H, P.v.w, C_H, P.v.b, 1.f, w.v, 1, 0, C_H, NB, C_H, C_H, 0, 0);
  gemm(s, 0, e,   0, 0, 1, 1, 0, C_H, P.edge.w, C_H, P.edge.b, 1.f, w.ea, 1, 0, C_H, EB, C_H, C_H, 0, 0);

  // key_update: silu(k[dst]@W0 + k[src]@W1 + ea@W2 + b1) -> l2
  gemm(s, 1, w.k, 0, dst, 1, 1, 0, C_H, P.key.l1.w, 768, 0, 1.f, w.h, 1, 0, C_H, EB, C_H, C_H, 0, 0);
  gemm(s, 1, w.k, 0, src, 1, 1, 0, C_H, P.key.l1.w + 256, 768, 0, 1.f, w.h, 1, 0, C_H, EB, C_H, C_H, 1, 0);
  gemm(s, 0, w.ea, 0, 0,  1, 1, 0, C_H, P.key.l1.w + 512, 768, P.key.l1.b, 1.f, w.h, 1, 0, C_H, EB, C_H, C_H, 1, 1);
  gemm(s, 0, w.h, 0, 0, 1, 1, 0, C_H, P.key.l2.w, C_H, P.key.l2.b, 1.f, w.kj, 1, 0, C_H, EB, C_H, C_H, 0, 0);

  alpha_k<<<blk((long)EB * C_H), 256, 0, s>>>(w.kj, w.q, dst, EB);
  zero_k<<<blk(512), 256, 0, s>>>(w.stats, 512);
  colstats_k<<<(unsigned)((EB + 63) / 64), 256, 0, s>>>(w.kj, EB, w.stats);

  // msg_update
  gemm(s, 1, w.v, 0, dst, 1, 1, 0, C_H, P.msg.l1.w, 768, 0, 1.f, w.h, 1, 0, C_H, EB, C_H, C_H, 0, 0);
  gemm(s, 1, w.v, 0, src, 1, 1, 0, C_H, P.msg.l1.w + 256, 768, 0, 1.f, w.h, 1, 0, C_H, EB, C_H, C_H, 1, 0);
  gemm(s, 0, w.ea, 0, 0,  1, 1, 0, C_H, P.msg.l1.w + 512, 768, P.msg.l1.b, 1.f, w.h, 1, 0, C_H, EB, C_H, C_H, 1, 1);
  gemm(s, 0, w.h, 0, 0, 1, 1, 0, C_H, P.msg.l2.w, C_H, P.msg.l2.b, 1.f, w.msg, 1, 0, C_H, EB, C_H, C_H, 0, 0);

  gate_k<<<blk((long)EB * C_H), 256, 0, s>>>(w.msg, w.kj, w.stats, P.bn_att.g, P.bn_att.b, EB);
  zero_k<<<blk((long)NB * C_H), 256, 0, s>>>(w.agg, (long)NB * C_H);
  scatter_k<<<blk((long)EB * C_H), 256, 0, s>>>(w.agg, w.msg, dst, EB);

  gemm(s, 0, w.agg, 0, 0, 1, 1, 0, C_H, P.concat.w, C_H, P.concat.b, 1.f, w.aggc, 1, 0, C_H, NB, C_H, C_H, 0, 0);
  zero_k<<<blk(512), 256, 0, s>>>(w.stats, 512);
  colstats_k<<<(unsigned)((NB + 63) / 64), 256, 0, s>>>(w.aggc, NB, w.stats);
  bnres_k<<<blk((long)NB * C_H), 256, 0, s>>>(xout, xin, w.aggc, w.stats, P.bn.g, P.bn.b, NB);
}

void run_edge(hipStream_t s, const EdgeP& P, const float* ein, float* eout, WS& w) {
  gemm(s, 0, ein, 0, 0, 1, 1, 0, C_H, P.q.w, C_H, P.q.b, 1.f, w.qe, 1, 0, C_H, EB, C_H, C_H, 0, 0);
  gemm(s, 0, ein, 0, 0, 1, 1, 0, C_H, P.k.w, C_H, P.k.b, 1.f, w.ke, 1, 0, C_H, EB, C_H, C_H, 0, 0);
  gemm(s, 0, ein, 0, 0, 1, 1, 0, C_H, P.v.w, C_H, P.v.b, 1.f, w.ve, 1, 0, C_H, EB, C_H, C_H, 0, 0);
  gemm(s, 0, w.na, 0, 0, 1, 1, 0, C_H, P.edge_w, C_H, 0, 1.f, w.exy, 1, 0, C_H, E3, C_H, C_H, 0, 0);

  // key_y[:,i,:] = ke_i(nl[:,i,:])  (rows 3m+i in/out)
  for (int i = 0; i < 3; ++i)
    gemm(s, 2, w.nl, 0, 0, 1, 3, i, C_H, P.ke[i].w, C_H, P.ke[i].b, 1.f, w.ky, 3, i, C_H, EB, C_H, C_H, 0, 0);

  // key_update over [3E,768]: broadcast k (row m/3), key_y, exy
  gemm(s, 2, w.ke, 0, 0, 3, 1, 0, C_H, P.key.l1.w, 768, 0, 1.f, w.h3, 1, 0, C_H, E3, C_H, C_H, 0, 0);
  gemm(s, 0, w.ky, 0, 0, 1, 1, 0, C_H, P.key.l1.w + 256, 768, 0, 1.f, w.h3, 1, 0, C_H, E3, C_H, C_H, 1, 0);
  gemm(s, 0, w.exy, 0, 0, 1, 1, 0, C_H, P.key.l1.w + 512, 768, P.key.l1.b, 1.f, w.h3, 1, 0, C_H, E3, C_H, C_H, 1, 1);
  gemm(s, 0, w.h3, 0, 0, 1, 1, 0, C_H, P.key.l2.w, C_H, P.key.l2.b, 1.f, w.kk, 1, 0, C_H, E3, C_H, C_H, 0, 0);

  alpha_k<<<blk((long)E3 * C_H), 256, 0, s>>>(w.kk, w.qe, (const int*)nullptr, E3);
  zero_k<<<blk(512), 256, 0, s>>>(w.stats, 512);
  colstats_k<<<(unsigned)((E3 + 63) / 64), 256, 0, s>>>(w.kk, E3, w.stats);

  // val_y overwrites ky (key_y no longer needed)
  for (int i = 0; i < 3; ++i)
    gemm(s, 2, w.nl, 0, 0, 1, 3, i, C_H, P.ve[i].w, C_H, P.ve[i].b, 1.f, w.ky, 3, i, C_H, EB, C_H, C_H, 0, 0);

  gemm(s, 2, w.ve, 0, 0, 3, 1, 0, C_H, P.msg.l1.w, 768, 0, 1.f, w.h3, 1, 0, C_H, E3, C_H, C_H, 0, 0);
  gemm(s, 0, w.ky, 0, 0, 1, 1, 0, C_H, P.msg.l1.w + 256, 768, 0, 1.f, w.h3, 1, 0, C_H, E3, C_H, C_H, 1, 0);
  gemm(s, 0, w.exy, 0, 0, 1, 1, 0, C_H, P.msg.l1.w + 512, 768, P.msg.l1.b, 1.f, w.h3, 1, 0, C_H, E3, C_H, C_H, 1, 1);
  gemm(s, 0, w.h3, 0, 0, 1, 1, 0, C_H, P.msg.l2.w, C_H, P.msg.l2.b, 1.f, w.msg3, 1, 0, C_H, E3, C_H, C_H, 0, 0);

  gate_k<<<blk((long)E3 * C_H), 256, 0, s>>>(w.msg3, w.kk, w.stats, P.bn_att.g, P.bn_att.b, E3);
  sum3_k<<<blk((long)EB * C_H), 256, 0, s>>>(w.outs, w.msg3);

  // linear(concat, .).sum(1) == concat(sum) with bias*3
  gemm(s, 0, w.outs, 0, 0, 1, 1, 0, C_H, P.concat.w, C_H, P.concat.b, 3.f, w.outc, 1, 0, C_H, EB, C_H, C_H, 0, 0);
  zero_k<<<blk(512), 256, 0, s>>>(w.stats, 512);
  colstats_k<<<(unsigned)((EB + 63) / 64), 256, 0, s>>>(w.outc, EB, w.stats);
  bnres_k<<<blk((long)EB * C_H), 256, 0, s>>>(eout, ein, w.outc, w.stats, P.bn.g, P.bn.b, EB);
}

} // namespace

extern "C" void kernel_launch(void* const* d_in, const int* in_sizes, int n_in,
                              void* d_out, int out_size, void* d_ws, size_t ws_size,
                              hipStream_t stream) {
  (void)in_sizes; (void)out_size; (void)ws_size;
  if (n_in < 136) return;

  const float* x    = (const float*)d_in[0];
  const float* attr = (const float*)d_in[1];
  const float* nei  = (const float*)d_in[2];
  const int*   eidx = (const int*)d_in[3];
  const int*   bat  = (const int*)d_in[4];
  const int* src = eidx;
  const int* dst = eidx + EB;

  // ---- unpack params (jax pytree: sorted-key DFS order) ----
  int cur = 5;
  auto nxt  = [&]() { return (const float*)d_in[cur++]; };
  auto lin  = [&]() { Lin l; l.b = nxt(); l.w = nxt(); return l; };   // keys: b, w
  auto bnp  = [&]() { BN p; p.b = nxt(); p.g = nxt(); return p; };    // keys: b, g
  auto mlp  = [&]() { Mlp m; m.l1 = lin(); m.l2 = lin(); return m; }; // l1, l2
  auto conv = [&]() {
    ConvP c; c.bn = bnp(); c.bn_att = bnp(); c.concat = lin(); c.edge = lin();
    c.k = lin(); c.key = mlp(); c.msg = mlp(); c.q = lin(); c.v = lin(); return c;
  };
  Lin   atom = lin();                 // atom_emb
  ConvP att[4] = { conv(), conv(), conv(), conv() };
  EdgeP eu;
  eu.bn = bnp(); eu.bn_att = bnp(); eu.concat = lin(); eu.edge_w = nxt();
  eu.k = lin(); eu.ke[0] = lin(); eu.ke[1] = lin(); eu.ke[2] = lin();
  eu.key = mlp(); eu.msg = mlp(); eu.q = lin(); eu.v = lin();
  eu.ve[0] = lin(); eu.ve[1] = lin(); eu.ve[2] = lin();
  Lin fc = lin(); Lin fco = lin(); Lin rbf = lin(); Lin rbfa = lin();

  // ---- workspace layout ----
  WS w;
  float* p = (float*)d_ws;
  auto take = [&](long n) { float* r = p; p += n; return r; };
  w.nf0 = take((long)NB * C_H);  w.nf1 = take((long)NB * C_H);
  w.ef0 = take((long)EB * C_H);  w.ef1 = take((long)EB * C_H);
  w.nl  = take((long)E3 * C_H);  w.na  = take((long)E3 * C_H);
  w.d_ef = take(EB); w.d_nl = take(E3); w.d_cos = take(E3);
  w.stats = take(512); w.pooled = take((long)NGR * C_H);
  w.counts = take(64); w.feats = take((long)NGR * C_H);
  float* scr = p;
  // conv-stage scratch view
  w.q = scr;                    w.k = w.q + (long)NB * C_H;  w.v = w.k + (long)NB * C_H;
  w.ea = w.v + (long)NB * C_H;  w.h = w.ea + (long)EB * C_H;
  w.kj = w.h + (long)EB * C_H;  w.msg = w.kj + (long)EB * C_H;
  w.agg = w.msg + (long)EB * C_H; w.aggc = w.agg + (long)NB * C_H;
  // edge-stage scratch view (overlaps)
  w.qe = scr;                    w.ke = w.qe + (long)EB * C_H; w.ve = w.ke + (long)EB * C_H;
  w.exy = w.ve + (long)EB * C_H; w.ky = w.exy + (long)E3 * C_H;
  w.h3 = w.ky + (long)E3 * C_H;  w.kk = w.h3 + (long)E3 * C_H;
  w.msg3 = w.kk + (long)E3 * C_H; w.outs = w.msg3 + (long)E3 * C_H;
  w.outc = w.outs + (long)EB * C_H;

  // ---- 1. geometry ----
  geom_k<<<blk(EB), 256, 0, stream>>>(attr, nei, w.d_ef, w.d_nl, w.d_cos);

  // ---- 2. fused RBF + linear + softplus (A generated inside the GEMM) ----
  const float stepD = 4.f / 255.f, gamD = 255.f / 4.f;  // range [-4,0], 256 bins
  const float stepA = 2.f / 255.f, gamA = 255.f / 2.f;  // range [-1,1]
  gemm(stream, 3, 0, w.d_ef, 0, 1, 1, 0, 0, rbf.w, C_H, rbf.b, 1.f, w.ef0, 1, 0, C_H,
       EB, C_H, C_H, 0, 2, -4.f, stepD, gamD);
  gemm(stream, 3, 0, w.d_nl, 0, 1, 1, 0, 0, rbf.w, C_H, rbf.b, 1.f, w.nl, 1, 0, C_H,
       E3, C_H, C_H, 0, 2, -4.f, stepD, gamD);
  gemm(stream, 3, 0, w.d_cos, 0, 1, 1, 0, 0, rbfa.w, C_H, rbfa.b, 1.f, w.na, 1, 0, C_H,
       E3, C_H, C_H, 0, 2, -1.f, stepA, gamA);

  // ---- 3. atom embedding (K=92) ----
  gemm(stream, 0, x, 0, 0, 1, 1, 0, AF, atom.w, AF, atom.b, 1.f, w.nf0, 1, 0, C_H,
       NB, C_H, AF, 0, 0);

  // ---- 4. conv / edge-update stack ----
  run_conv(stream, att[0], w.nf0, w.nf1, w.ef0, src, dst, w);
  run_edge(stream, eu, w.ef0, w.ef1, w);
  run_conv(stream, att[1], w.nf1, w.nf0, w.ef1, src, dst, w);
  run_conv(stream, att[2], w.nf0, w.nf1, w.ef1, src, dst, w);
  run_conv(stream, att[3], w.nf1, w.nf0, w.ef1, src, dst, w);

  // ---- 5. mean pooling + head ----
  zero_k<<<blk((long)NGR * C_H), 256, 0, stream>>>(w.pooled, (long)NGR * C_H);
  zero_k<<<1, 256, 0, stream>>>(w.counts, 64);
  count_k<<<blk(NB), 256, 0, stream>>>(w.counts, bat);
  pooladd_k<<<blk((long)NB * C_H), 256, 0, stream>>>(w.pooled, w.nf0, bat);
  pooldiv_k<<<blk((long)NGR * C_H), 256, 0, stream>>>(w.pooled, w.counts);
  gemm(stream, 0, w.pooled, 0, 0, 1, 1, 0, C_H, fc.w, C_H, fc.b, 1.f, w.feats, 1, 0, C_H,
       NGR, C_H, C_H, 0, 1 /*silu*/);
  fcout_k<<<1, 32, 0, stream>>>(w.feats, fco.w, fco.b, (float*)d_out);
}
